// LSTMPredictor_65481071395055
// MI455X (gfx1250) — matrix operand order repaired and checked by reading, not compiled, not run on hardware
//
#include <hip/hip_runtime.h>

#define B_TOT 16384
#define S_LEN 64
#define T_LEN 64
#define I_DIM 4
#define M_DIM 16
#define H_DIM 64

#define NWAVE 4
#define NTHREADS (NWAVE * 32)
#define ROWS_PER_WAVE 16
#define ROWS_PER_WG (NWAVE * ROWS_PER_WAVE)

typedef __attribute__((ext_vector_type(16))) _Float16 v16h;
typedef __attribute__((ext_vector_type(8)))  _Float16 v8h;
typedef __attribute__((ext_vector_type(8)))  float    v8f;

__device__ __forceinline__ float sigm(float x) { return 1.0f / (1.0f + __expf(-x)); }
__device__ __forceinline__ float tanh_fast(float x) { return 2.0f / (1.0f + __expf(-2.0f * x)) - 1.0f; }

// B operand (K x 16), f16 32x16 tile. Weight stored row-major W[n][k] (N=256 rows, K cols).
// Lane l: column n = ntile*16 + (l&15); halves[0..7] = K kc*32 + (l>>4)*8 .., halves[8..15] = +16.
__device__ __forceinline__ v16h load_B_tile(const _Float16* W, int K, int ntile, int kc, int lane) {
  const _Float16* p = W + (ntile * 16 + (lane & 15)) * K + kc * 32 + (lane >> 4) * 8;
  v16h r;
  *((v8h*)&r)       = *(const v8h*)(p);
  *(((v8h*)&r) + 1) = *(const v8h*)(p + 16);
  return r;
}

// A operand (16 x 32), f16. Source row-major A[m][k], stride K.
// Lane l: row m = l&15; halves[0..7] = K (l>>4)*8 + kc*32 .., halves[8..15] = +16.
__device__ __forceinline__ v16h load_A_tile(const _Float16* A, int K, int kc, int lane) {
  const _Float16* p = A + (lane & 15) * K + kc * 32 + (lane >> 4) * 8;
  v16h r;
  *((v8h*)&r)       = *(const v8h*)(p);
  *(((v8h*)&r) + 1) = *(const v8h*)(p + 16);
  return r;
}

// One LSTM cell step for a wave's 16 batch rows.
// gates(16x256) = xA(16xXK) @ Wih^T + h(16x64) @ Whh^T + bias, then elementwise update.
// hbuf (16x64 f16, row-major) is both the recurrent A input and the output.
// WMMA ordering: k-chunk outer, gate inner -> 4 independent accumulator chains.
// A sched_barrier at the jt=1/jt=2 boundary caps accumulator liveness at 2 jt
// groups (keeps VGPR allocation under 256 so we stay at 2 waves/SIMD).
template <int XK>
__device__ __forceinline__ void lstm_cell(
    const _Float16* __restrict__ xA, const _Float16* __restrict__ Wih,
    _Float16* __restrict__ hbuf, const _Float16* __restrict__ Whh,
    const float* __restrict__ bias,
    float (&c)[4][8], int lane)
{
  const int lmod = lane & 15, lhi = lane >> 4;
  __syncthreads();  // writers of xA / hbuf are done
  v16h ax0 = load_A_tile(xA, XK, 0, lane);
  v16h ax1{};
  if (XK == 64) ax1 = load_A_tile(xA, 64, 1, lane);
  v16h ah0 = load_A_tile(hbuf, 64, 0, lane);
  v16h ah1 = load_A_tile(hbuf, 64, 1, lane);
  __syncthreads();  // all lanes captured h before we overwrite it

#pragma unroll
  for (int jt = 0; jt < 4; ++jt) {
    // batched bias loads for this jt group (one clause, one wait)
    float bv[4];
#pragma unroll
    for (int g = 0; g < 4; ++g) bv[g] = bias[g * 64 + jt * 16 + lmod];
    v8f acc[4];
#pragma unroll
    for (int g = 0; g < 4; ++g) {
#pragma unroll
      for (int v = 0; v < 8; ++v) acc[g][v] = bv[g];
    }

#pragma unroll
    for (int g = 0; g < 4; ++g)
      acc[g] = __builtin_amdgcn_wmma_f32_16x16x32_f16(
          false, ax0, false, load_B_tile(Wih, XK, g * 4 + jt, 0, lane),
          (short)0, acc[g], false, false);
    if (XK == 64) {
#pragma unroll
      for (int g = 0; g < 4; ++g)
        acc[g] = __builtin_amdgcn_wmma_f32_16x16x32_f16(
            false, ax1, false, load_B_tile(Wih, 64, g * 4 + jt, 1, lane),
            (short)0, acc[g], false, false);
    }
#pragma unroll
    for (int g = 0; g < 4; ++g)
      acc[g] = __builtin_amdgcn_wmma_f32_16x16x32_f16(
          false, ah0, false, load_B_tile(Whh, 64, g * 4 + jt, 0, lane),
          (short)0, acc[g], false, false);
#pragma unroll
    for (int g = 0; g < 4; ++g)
      acc[g] = __builtin_amdgcn_wmma_f32_16x16x32_f16(
          false, ah1, false, load_B_tile(Whh, 64, g * 4 + jt, 1, lane),
          (short)0, acc[g], false, false);

    // D layout: lane holds column n = jt*16+lmod, rows m = lhi*8 + v
#pragma unroll
    for (int v = 0; v < 8; ++v) {
      float ig = sigm(acc[0][v]);
      float fg = sigm(acc[1][v]);
      float gg = tanh_fast(acc[2][v]);
      float og = sigm(acc[3][v]);
      float cn = fg * c[jt][v] + ig * gg;
      c[jt][v] = cn;
      float hn = og * tanh_fast(cn);
      hbuf[(lhi * 8 + v) * 64 + jt * 16 + lmod] = (_Float16)hn;
    }
    if (jt == 1) __builtin_amdgcn_sched_barrier(0);  // cap live accumulator groups
  }
}

__device__ void pack_w(_Float16* dst, const float* src, int rows, int ksrc, int kdst, int tid) {
  int n = rows * kdst;
  for (int idx = tid; idx < n; idx += NTHREADS) {
    int r = idx / kdst, k = idx - r * kdst;
    dst[idx] = (_Float16)((k < ksrc) ? src[r * ksrc + k] : 0.0f);
  }
}
__device__ void copy_f(float* dst, const float* src, int n, int tid) {
  for (int i = tid; i < n; i += NTHREADS) dst[i] = src[i];
}

__global__ __launch_bounds__(NTHREADS, 1) __attribute__((amdgpu_waves_per_eu(2)))
void lstm_seq2seq_kernel(
    const float* __restrict__ src, const float* __restrict__ trg,
    const float* __restrict__ W_in, const float* __restrict__ b_in,
    const float* __restrict__ eWih0, const float* __restrict__ eWhh0, const float* __restrict__ eb0,
    const float* __restrict__ eWih1, const float* __restrict__ eWhh1, const float* __restrict__ eb1,
    const float* __restrict__ dWih0, const float* __restrict__ dWhh0, const float* __restrict__ db0,
    const float* __restrict__ dWih1, const float* __restrict__ dWhh1, const float* __restrict__ db1,
    const float* __restrict__ W_out, const float* __restrict__ b_out,
    float* __restrict__ out)
{
  // Phase-shared weight storage (encoder first, then repacked with decoder weights)
  __shared__ _Float16 sWih0[256 * 32];   // K=16 padded to 32
  __shared__ _Float16 sWhh0[256 * 64];
  __shared__ _Float16 sWih1[256 * 64];
  __shared__ _Float16 sWhh1[256 * 64];
  __shared__ float sB0[256], sB1[256];
  __shared__ float sWin[64], sBin[16], sWout[256], sBout[4];
  __shared__ _Float16 sX[NWAVE][16 * 32];   // A operand for layer0 input (K padded to 32)
  __shared__ _Float16 sH1[NWAVE][16 * 64];  // layer0 hidden state, row-major
  __shared__ _Float16 sH2[NWAVE][16 * 64];  // layer1 hidden state, row-major
  __shared__ float sXin[NWAVE][64];         // decoder carry x (16 rows x 4)

  const int tid = threadIdx.x;
  const int wave = tid >> 5, lane = tid & 31;
  const int lmod = lane & 15, lhi = lane >> 4;
  const long base = (long)blockIdx.x * ROWS_PER_WG + wave * ROWS_PER_WAVE;

  // ---- pack encoder weights into LDS (f16, B-operand friendly row-major) ----
  pack_w(sWih0, eWih0, 256, 16, 32, tid);
  pack_w(sWhh0, eWhh0, 256, 64, 64, tid);
  pack_w(sWih1, eWih1, 256, 64, 64, tid);
  pack_w(sWhh1, eWhh1, 256, 64, 64, tid);
  copy_f(sB0, eb0, 256, tid);
  copy_f(sB1, eb1, 256, tid);
  copy_f(sWin, W_in, 64, tid);
  copy_f(sBin, b_in, 16, tid);
  copy_f(sWout, W_out, 256, tid);
  copy_f(sBout, b_out, 4, tid);
  for (int i = tid; i < NWAVE * 16 * 32; i += NTHREADS) (&sX[0][0])[i] = (_Float16)0.0f;
  for (int i = tid; i < NWAVE * 16 * 64; i += NTHREADS) {
    (&sH1[0][0])[i] = (_Float16)0.0f;
    (&sH2[0][0])[i] = (_Float16)0.0f;
  }
  __syncthreads();

  float c1[4][8], c2[4][8];
#pragma unroll
  for (int jt = 0; jt < 4; ++jt)
#pragma unroll
    for (int v = 0; v < 8; ++v) { c1[jt][v] = 0.0f; c2[jt][v] = 0.0f; }

  _Float16* xw  = sX[wave];
  _Float16* h1w = sH1[wave];
  _Float16* h2w = sH2[wave];
  float*    xinw = sXin[wave];

  // ================= encoder =================
  for (int t = 0; t < S_LEN; ++t) {
    // mid(16x16) = src_t @ W_in^T + b_in, written in A row-major (pad cols 16..31 stay 0)
#pragma unroll
    for (int v = 0; v < 8; ++v) {
      int r = lhi * 8 + v;
      const float* sp = src + ((base + r) * S_LEN + t) * I_DIM;
      float acc = sBin[lmod];
#pragma unroll
      for (int k = 0; k < 4; ++k) acc += sp[k] * sWin[lmod * 4 + k];
      xw[r * 32 + lmod] = (_Float16)acc;
    }
    lstm_cell<32>(xw, sWih0, h1w, sWhh0, sB0, c1, lane);
    lstm_cell<64>(h1w, sWih1, h2w, sWhh1, sB1, c2, lane);
  }

  // ---- repack with decoder weights (states in sH1/sH2/c regs carry over) ----
  __syncthreads();
  pack_w(sWih0, dWih0, 256, 16, 32, tid);
  pack_w(sWhh0, dWhh0, 256, 64, 64, tid);
  pack_w(sWih1, dWih1, 256, 64, 64, tid);
  pack_w(sWhh1, dWhh1, 256, 64, 64, tid);
  copy_f(sB0, db0, 256, tid);
  copy_f(sB1, db1, 256, tid);

  // decoder initial carry: x = trg[:, 0]
  {
#pragma unroll
    for (int q = 0; q < 2; ++q) {
      int p = lane + q * 32;            // 64 items = 16 rows x 4
      int r = p >> 2, ii = p & 3;
      xinw[p] = trg[((base + r) * T_LEN + 0) * I_DIM + ii];
    }
  }
  __syncthreads();

  // ================= decoder =================
  for (int t = 0; t < T_LEN; ++t) {
    // mid from carried x
#pragma unroll
    for (int v = 0; v < 8; ++v) {
      int r = lhi * 8 + v;
      float acc = sBin[lmod];
#pragma unroll
      for (int k = 0; k < 4; ++k) acc += xinw[r * 4 + k] * sWin[lmod * 4 + k];
      xw[r * 32 + lmod] = (_Float16)acc;
    }
    lstm_cell<32>(xw, sWih0, h1w, sWhh0, sB0, c1, lane);
    lstm_cell<64>(h1w, sWih1, h2w, sWhh1, sB1, c2, lane);
    __syncthreads();  // h2w writes visible

    // pred = h2 @ W_out^T + b_out ; out[:,t] = pred + trg[:,t] ; carry = same value
#pragma unroll
    for (int q = 0; q < 2; ++q) {
      int p = lane + q * 32;
      int r = p >> 2, ii = p & 3;
      float acc = sBout[ii];
#pragma unroll 8
      for (int j = 0; j < 64; ++j)
        acc += (float)h2w[r * 64 + j] * sWout[ii * 64 + j];
      float o = acc + trg[((base + r) * T_LEN + t) * I_DIM + ii];
      out[((base + r) * T_LEN + t) * I_DIM + ii] = o;
      xinw[p] = o;
    }
    __syncthreads();  // xinw ready for next step's mid
  }
}

extern "C" void kernel_launch(void* const* d_in, const int* in_sizes, int n_in,
                              void* d_out, int out_size, void* d_ws, size_t ws_size,
                              hipStream_t stream) {
  (void)in_sizes; (void)n_in; (void)out_size; (void)d_ws; (void)ws_size;
  const float* src   = (const float*)d_in[0];
  const float* trg   = (const float*)d_in[1];
  const float* W_in  = (const float*)d_in[2];
  const float* b_in  = (const float*)d_in[3];
  const float* eWih0 = (const float*)d_in[4];
  const float* eWhh0 = (const float*)d_in[5];
  const float* eb0   = (const float*)d_in[6];
  const float* eWih1 = (const float*)d_in[7];
  const float* eWhh1 = (const float*)d_in[8];
  const float* eb1   = (const float*)d_in[9];
  const float* dWih0 = (const float*)d_in[10];
  const float* dWhh0 = (const float*)d_in[11];
  const float* db0   = (const float*)d_in[12];
  const float* dWih1 = (const float*)d_in[13];
  const float* dWhh1 = (const float*)d_in[14];
  const float* db1   = (const float*)d_in[15];
  const float* W_out = (const float*)d_in[16];
  const float* b_out = (const float*)d_in[17];
  float* out = (float*)d_out;

  dim3 grid(B_TOT / ROWS_PER_WG);
  dim3 block(NTHREADS);
  lstm_seq2seq_kernel<<<grid, block, 0, stream>>>(
      src, trg, W_in, b_in,
      eWih0, eWhh0, eb0, eWih1, eWhh1, eb1,
      dWih0, dWhh0, db0, dWih1, dWhh1, db1,
      W_out, b_out, out);
}
